// DecoderLayer_86165633892726
// MI455X (gfx1250) — matrix operand order repaired and checked by reading, not compiled, not run on hardware
//
#include <hip/hip_runtime.h>

// ---------------------------------------------------------------------------
// Decoder layer for gfx1250 (MI455X): bf16 WMMA GEMMs + fused attention,
// async global->LDS staging (ASYNCcnt) with LDS double buffering.
// ---------------------------------------------------------------------------

typedef unsigned short ushort_t;
typedef __attribute__((ext_vector_type(16))) __bf16 v16bf;
typedef __attribute__((ext_vector_type(8)))  float  v8f;

#define DEV static __device__ __forceinline__

DEV ushort_t f2bf(float f) {
    union { float f; unsigned u; } v; v.f = f;
    unsigned u = v.u;
    unsigned r = 0x7FFFu + ((u >> 16) & 1u);
    return (ushort_t)((u + r) >> 16);
}

DEV unsigned pack2(float a, float b) {
    return (unsigned)f2bf(a) | ((unsigned)f2bf(b) << 16);
}

DEV v8f wmma_bf16(v16bf a, v16bf b, v8f c) {
    return __builtin_amdgcn_wmma_f32_16x16x32_bf16(
        false, a, false, b, (short)0, c, false, false);
}

union FragU { uint4 u[2]; v16bf v; };

// load a 16-element bf16 fragment: 8 elems at p, 8 elems at p+gapElems
DEV v16bf ldfrag(const ushort_t* p, int gapElems) {
    FragU t;
    t.u[0] = *(const uint4*)p;
    t.u[1] = *(const uint4*)(p + gapElems);
    return t.v;
}

// ---- CDNA5 async global->LDS copy (16B per lane), tracked by ASYNCcnt -----
typedef int v4i_vs __attribute__((vector_size(16)));
typedef __attribute__((address_space(1))) v4i_vs* g_v4i_ptr;
typedef __attribute__((address_space(3))) v4i_vs* l_v4i_ptr;

DEV void async_copy_b128(const ushort_t* g, ushort_t* l) {
#if __has_builtin(__builtin_amdgcn_global_load_async_to_lds_b128)
    __builtin_amdgcn_global_load_async_to_lds_b128(
        (g_v4i_ptr)(size_t)g, (l_v4i_ptr)(size_t)l, 0, 0);
#else
    unsigned loff = (unsigned)(size_t)(void*)l;  // LDS aperture: low 32 bits
    asm volatile("global_load_async_to_lds_b128 %0, %1, off"
                 :: "v"(loff), "v"(g)
                 : "memory");
#endif
}

template <int N>
DEV void wait_async() {
#if __has_builtin(__builtin_amdgcn_s_wait_asynccnt)
    __builtin_amdgcn_s_wait_asynccnt((unsigned short)N);
#else
    asm volatile("s_wait_asynccnt %0" :: "i"(N) : "memory");
#endif
}

// ---------------------------------------------------------------------------
// elementwise f32 -> bf16 convert (vectorized x4)
// ---------------------------------------------------------------------------
__global__ __launch_bounds__(256)
void cvt_f32_bf16_kernel(const float4* __restrict__ in, uint2* __restrict__ out,
                         long long n4) {
    long long stride = (long long)gridDim.x * blockDim.x;
    for (long long i = (long long)blockIdx.x * blockDim.x + threadIdx.x;
         i < n4; i += stride) {
        float4 v = in[i];
        uint2 o;
        o.x = pack2(v.x, v.y);
        o.y = pack2(v.z, v.w);
        out[i] = o;
    }
}

// ---------------------------------------------------------------------------
// W[K][N] f32 (row-major) -> WT[N][K] bf16 (row-major)  (32x32 LDS tiles)
// ---------------------------------------------------------------------------
__global__ __launch_bounds__(256)
void transpose_cvt_kernel(const float* __restrict__ W, ushort_t* __restrict__ WT,
                          int K, int N) {
    __shared__ float t[32][33];
    const int tx = threadIdx.x & 31, ty = threadIdx.x >> 5;
    const int n0 = blockIdx.x * 32, k0 = blockIdx.y * 32;
#pragma unroll
    for (int i = 0; i < 4; ++i) {
        int r = ty + i * 8;
        t[r][tx] = W[(size_t)(k0 + r) * N + n0 + tx];
    }
    __syncthreads();
#pragma unroll
    for (int i = 0; i < 4; ++i) {
        int r = ty + i * 8;
        WT[(size_t)(n0 + r) * K + k0 + tx] = f2bf(t[tx][r]);
    }
}

// ---------------------------------------------------------------------------
// Universal bf16 GEMM:  C[M,N] = A[M,K] * Bt[N,K]^T  (+bias, +relu)
//   BM=128, BK=32, BN template {128,64}; 256 threads = 8 wave32 (4x2 grid)
//   async global->LDS staging with double-buffered tiles.
//   batched via blockIdx.z: b = z>>4, h = z&15 offsets (0 for single batch)
//   outputs: Cf (f32), Cb (bf16), CbT (bf16, [b][n][s] "V-transposed" layout)
// ---------------------------------------------------------------------------
template <int BN>
__global__ __launch_bounds__(256)
void gemm_bf16_kernel(const ushort_t* __restrict__ A, int lda, long long as0, long long as1,
                      const ushort_t* __restrict__ Bt, int ldb, long long bs0, long long bs1,
                      float* __restrict__ Cf, ushort_t* __restrict__ Cb,
                      ushort_t* __restrict__ CbT,
                      int ldc, long long cs0, long long cs1,
                      const float* __restrict__ bias, int relu, int K,
                      int tS, long long tBstr) {
    constexpr int BM = 128, BK = 32;
    constexpr int NF = BN / 32;                 // n-frags per wave
    constexpr int ACH = (BM * BK) / (256 * 8);  // async ops/thread for A tile
    constexpr int BCH = (BN * BK) / (256 * 8);  // async ops/thread for B tile
    constexpr int TOPS = ACH + BCH;             // per-wave ASYNCcnt per tile
    __shared__ __align__(16) ushort_t As[2][BM * BK];
    __shared__ __align__(16) ushort_t Bs[2][BN * BK];

    const int tid = threadIdx.x;
    const int lane = tid & 31, l15 = lane & 15, half = lane >> 4;
    const int wv = tid >> 5, wm = wv & 3, wn = wv >> 2;
    const int z = blockIdx.z, zb = z >> 4, zh = z & 15;
    const int m0 = blockIdx.x * BM, n0 = blockIdx.y * BN;

    const ushort_t* Ag = A + (size_t)zb * as0 + (size_t)zh * as1;
    const ushort_t* Bg = Bt + (size_t)zb * bs0 + (size_t)zh * bs1;

    v8f acc[2][NF];
#pragma unroll
    for (int fm = 0; fm < 2; ++fm)
#pragma unroll
        for (int fn = 0; fn < NF; ++fn) acc[fm][fn] = (v8f){};

    auto stage = [&](int k0, int buf) {
#pragma unroll
        for (int c = 0; c < ACH; ++c) {
            int flat = (tid + c * 256) * 8;
            int r = flat >> 5, cc = flat & 31;
            async_copy_b128(Ag + (size_t)(m0 + r) * lda + k0 + cc, &As[buf][flat]);
        }
#pragma unroll
        for (int c = 0; c < BCH; ++c) {
            int flat = (tid + c * 256) * 8;
            int r = flat >> 5, cc = flat & 31;
            async_copy_b128(Bg + (size_t)(n0 + r) * ldb + k0 + cc, &Bs[buf][flat]);
        }
    };

    const int nk = K / BK;
    stage(0, 0);
    for (int i = 0; i < nk; ++i) {
        const int cur = i & 1;
        if (i + 1 < nk) {
            stage((i + 1) * BK, cur ^ 1);  // prefetch next tile into other buf
            wait_async<TOPS>();            // drain THIS tile's copies only
        } else {
            wait_async<0>();
        }
        __syncthreads();

        // ---- A fragments (ISA 16-bit A 16x32 layout) ----
        FragU af[2];
        const int kb = half * 8;
#pragma unroll
        for (int fm = 0; fm < 2; ++fm) {
            int r = wm * 32 + fm * 16 + l15;
            af[fm].u[0] = *(const uint4*)&As[cur][r * BK + kb];
            af[fm].u[1] = *(const uint4*)&As[cur][r * BK + kb + 16];
        }
        const int kb2 = half * 8;
#pragma unroll
        for (int fn = 0; fn < NF; ++fn) {
            int n = wn * (BN / 2) + fn * 16 + l15;
            FragU bfv;
            bfv.u[0] = *(const uint4*)&Bs[cur][n * BK + kb2];
            bfv.u[1] = *(const uint4*)&Bs[cur][n * BK + kb2 + 16];
#pragma unroll
            for (int fm = 0; fm < 2; ++fm)
                acc[fm][fn] = wmma_bf16(af[fm].v, bfv.v, acc[fm][fn]);
        }
        __syncthreads();  // all reads done before buffer is overwritten
    }

    // ---- epilogue: bias / relu / stores (C layout: row=j+8*half, col=l15) --
    const size_t cb = (size_t)zb * cs0 + (size_t)zh * cs1;
#pragma unroll
    for (int fm = 0; fm < 2; ++fm) {
#pragma unroll
        for (int fn = 0; fn < NF; ++fn) {
            int nn = n0 + wn * (BN / 2) + fn * 16 + l15;
            float bv = bias ? bias[nn] : 0.f;
            int rb = m0 + wm * 32 + fm * 16 + 8 * half;
#pragma unroll
            for (int j = 0; j < 8; ++j) {
                float v = acc[fm][fn][j] + bv;
                if (relu) v = fmaxf(v, 0.f);
                int mm = rb + j;
                if (Cf) Cf[cb + (size_t)mm * ldc + nn] = v;
                if (Cb) Cb[cb + (size_t)mm * ldc + nn] = f2bf(v);
                if (CbT) {
                    int bb = mm / tS, ss = mm - bb * tS;
                    CbT[(size_t)bb * tBstr + (size_t)nn * tS + ss] = f2bf(v);
                }
            }
        }
    }
}

// ---------------------------------------------------------------------------
// Fused attention logits + mask + softmax -> bf16 probs (and optional f32
// logits for the attn_logits2 output). One block = 16 query rows x full Skv.
// Q,KT: [B, S, 1024] bf16 (per b,h slice at offset b*S*1024 + h*64), depth 64.
// grid: (Sq/16, 1, B*H); block 256 = 8 waves, each wave owns 256 columns.
// B-fragments are software-pipelined 2 deep to hide global latency.
// ---------------------------------------------------------------------------
__global__ __launch_bounds__(256)
void attn_softmax_kernel(const ushort_t* __restrict__ Q, const ushort_t* __restrict__ KT,
                         ushort_t* __restrict__ P, float* __restrict__ L,
                         const float* __restrict__ mask, int maskMode,
                         int Sq, int Skv, float scale) {
    __shared__ float red[8][16];
    __shared__ float gmax[16];
    __shared__ float gsum[16];
    const int tid = threadIdx.x, lane = tid & 31, wv = tid >> 5;
    const int l15 = lane & 15, half = lane >> 4;
    const int z = blockIdx.z, b = z >> 4, h = z & 15;
    const int m0 = blockIdx.x * 16;
    const size_t qbase = (size_t)b * Sq * 1024 + h * 64;
    const size_t kbase = (size_t)b * Skv * 1024 + h * 64;
    const size_t obase = (size_t)z * Sq * Skv;

    // A fragments: depth K=64 -> two 16x32 frags, straight from global
    const ushort_t* ap = Q + qbase + (size_t)(m0 + l15) * 1024 + half * 16;
    v16bf a0 = ldfrag(ap, 8);
    v16bf a1 = ldfrag(ap + 32, 8);

    const int ncol0 = wv * 256;
    const ushort_t* bpBase = KT + kbase + half * 16;

    v8f acc[16];
    v16bf pb0[2], pb1[2];
    {
        const ushort_t* p = bpBase + (size_t)(ncol0 + l15) * 1024;
        pb0[0] = ldfrag(p, 8);
        pb1[0] = ldfrag(p + 32, 8);
    }
#pragma unroll
    for (int nf = 0; nf < 16; ++nf) {
        if (nf + 1 < 16) {  // prefetch next column-tile's fragments
            const ushort_t* p = bpBase + (size_t)(ncol0 + (nf + 1) * 16 + l15) * 1024;
            pb0[(nf + 1) & 1] = ldfrag(p, 8);
            pb1[(nf + 1) & 1] = ldfrag(p + 32, 8);
        }
        v8f c = (v8f){};
        c = wmma_bf16(a0, pb0[nf & 1], c);
        c = wmma_bf16(a1, pb1[nf & 1], c);
        acc[nf] = c;
    }

    // scale + mask (+ raw logits output)
#pragma unroll
    for (int nf = 0; nf < 16; ++nf) {
        int n = ncol0 + nf * 16 + l15;
#pragma unroll
        for (int j = 0; j < 8; ++j) {
            int m = m0 + j + 8 * half;
            float v = acc[nf][j] * scale;
            if (maskMode == 1)
                v += mask[(size_t)b * Sq * Skv + (size_t)m * Skv + n] * (-1e9f);
            else if (maskMode == 2)
                v += mask[(size_t)b * Skv + n] * (-1e9f);
            acc[nf][j] = v;
            if (L) L[obase + (size_t)m * Skv + n] = v;
        }
    }

    // ---- row max (16 lanes -> wave -> block) ----
    float mx[8];
#pragma unroll
    for (int j = 0; j < 8; ++j) {
        float v = -3.0e38f;
#pragma unroll
        for (int nf = 0; nf < 16; ++nf) v = fmaxf(v, acc[nf][j]);
        for (int s = 1; s < 16; s <<= 1) v = fmaxf(v, __shfl_xor(v, s, 32));
        mx[j] = v;
    }
    if (l15 == 0) {
#pragma unroll
        for (int j = 0; j < 8; ++j) red[wv][j + 8 * half] = mx[j];
    }
    __syncthreads();
    if (tid < 16) {
        float v = red[0][tid];
#pragma unroll
        for (int w = 1; w < 8; ++w) v = fmaxf(v, red[w][tid]);
        gmax[tid] = v;
    }
    __syncthreads();
    float rmax[8];
#pragma unroll
    for (int j = 0; j < 8; ++j) rmax[j] = gmax[j + 8 * half];

    // ---- exp + row sum ----
    float sm[8];
#pragma unroll
    for (int j = 0; j < 8; ++j) {
        float s = 0.f;
#pragma unroll
        for (int nf = 0; nf < 16; ++nf) {
            float e = __expf(acc[nf][j] - rmax[j]);
            acc[nf][j] = e;
            s += e;
        }
        for (int m = 1; m < 16; m <<= 1) s += __shfl_xor(s, m, 32);
        sm[j] = s;
    }
    if (l15 == 0) {
#pragma unroll
        for (int j = 0; j < 8; ++j) red[wv][j + 8 * half] = sm[j];
    }
    __syncthreads();
    if (tid < 16) {
        float v = 0.f;
#pragma unroll
        for (int w = 0; w < 8; ++w) v += red[w][tid];
        gsum[tid] = v;
    }
    __syncthreads();
    float rinv[8];
#pragma unroll
    for (int j = 0; j < 8; ++j) rinv[j] = 1.f / gsum[j + 8 * half];

    // ---- store bf16 probs ----
#pragma unroll
    for (int nf = 0; nf < 16; ++nf) {
        int n = ncol0 + nf * 16 + l15;
#pragma unroll
        for (int j = 0; j < 8; ++j) {
            int m = m0 + j + 8 * half;
            P[obase + (size_t)m * Skv + n] = f2bf(acc[nf][j] * rinv[j]);
        }
    }
}

// ---------------------------------------------------------------------------
// LayerNorm over D=1024 with fused residual: y = LN(X + R) * g + b
// One 256-thread block per row; outputs f32 (+ optional bf16).
// ---------------------------------------------------------------------------
DEV float block_sum256(float v, float* sh) {
    for (int m = 16; m >= 1; m >>= 1) v += __shfl_xor(v, m, 32);
    int w = threadIdx.x >> 5;
    if ((threadIdx.x & 31) == 0) sh[w] = v;
    __syncthreads();
    if (threadIdx.x < 8) {
        float s = sh[threadIdx.x];
        for (int m = 4; m >= 1; m >>= 1) s += __shfl_xor(s, m, 32);
        if (threadIdx.x == 0) sh[0] = s;
    }
    __syncthreads();
    float r = sh[0];
    __syncthreads();
    return r;
}

__global__ __launch_bounds__(256)
void layernorm_kernel(const float* __restrict__ X, const float* __restrict__ R,
                      const float* __restrict__ g, const float* __restrict__ beta,
                      float* __restrict__ outF, ushort_t* __restrict__ outB) {
    __shared__ float sh[8];
    const int t = threadIdx.x;
    const size_t base = (size_t)blockIdx.x * 1024;
    float4 x = ((const float4*)(X + base))[t];
    if (R) {
        float4 r = ((const float4*)(R + base))[t];
        x.x += r.x; x.y += r.y; x.z += r.z; x.w += r.w;
    }
    float s = block_sum256(x.x + x.y + x.z + x.w, sh);
    float mu = s * (1.f / 1024.f);
    float d0 = x.x - mu, d1 = x.y - mu, d2 = x.z - mu, d3 = x.w - mu;
    float ss = block_sum256(d0 * d0 + d1 * d1 + d2 * d2 + d3 * d3, sh);
    float rstd = rsqrtf(ss * (1.f / 1024.f) + 1e-6f);
    float4 g4 = ((const float4*)g)[t];
    float4 b4 = ((const float4*)beta)[t];
    float4 y;
    y.x = d0 * rstd * g4.x + b4.x;
    y.y = d1 * rstd * g4.y + b4.y;
    y.z = d2 * rstd * g4.z + b4.z;
    y.w = d3 * rstd * g4.w + b4.w;
    ((float4*)(outF + base))[t] = y;
    if (outB) {
        uint2 o;
        o.x = pack2(y.x, y.y);
        o.y = pack2(y.z, y.w);
        ((uint2*)(outB + base))[t] = o;
    }
}

// ---------------------------------------------------------------------------
// host orchestration
// ---------------------------------------------------------------------------
extern "C" void kernel_launch(void* const* d_in, const int* in_sizes, int n_in,
                              void* d_out, int out_size, void* d_ws, size_t ws_size,
                              hipStream_t stream) {
    (void)in_sizes; (void)n_in; (void)out_size; (void)ws_size;
    constexpr int Bn = 4, S = 2048, SE = 2048, D = 1024, H = 16, DFF = 4096;
    constexpr long long M = (long long)Bn * S;            // 8192 rows
    constexpr long long ME = M * D;                       // activation elems
    constexpr long long PE = (long long)Bn * H * S * SE;  // probs / logits elems
    constexpr long long HE = M * DFF;                     // ffn hidden elems

    const float* x    = (const float*)d_in[0];
    const float* enc  = (const float*)d_in[1];
    const float* lam  = (const float*)d_in[2];
    const float* pam  = (const float*)d_in[3];
    const float* wq1 = (const float*)d_in[4];  const float* bq1 = (const float*)d_in[5];
    const float* wk1 = (const float*)d_in[6];  const float* bk1 = (const float*)d_in[7];
    const float* wv1 = (const float*)d_in[8];  const float* bv1 = (const float*)d_in[9];
    const float* wo1 = (const float*)d_in[10]; const float* bo1 = (const float*)d_in[11];
    const float* wq2 = (const float*)d_in[12]; const float* bq2 = (const float*)d_in[13];
    const float* wk2 = (const float*)d_in[14]; const float* bk2 = (const float*)d_in[15];
    const float* wv2 = (const float*)d_in[16]; const float* bv2 = (const float*)d_in[17];
    const float* wo2 = (const float*)d_in[18]; const float* bo2 = (const float*)d_in[19];
    const float* wf1 = (const float*)d_in[20]; const float* bf1 = (const float*)d_in[21];
    const float* wf2 = (const float*)d_in[22]; const float* bf2 = (const float*)d_in[23];
    const float* ln2g = (const float*)d_in[24]; const float* ln2b = (const float*)d_in[25];
    const float* ln3g = (const float*)d_in[26]; const float* ln3b = (const float*)d_in[27];

    float* out3   = (float*)d_out;
    float* logits = (float*)d_out + ME;  // attn_logits2 [B,H,S,SE]

    // ---- workspace bump allocator ----
    char* wsb = (char*)d_ws;
    size_t off = 0;
    auto alloc = [&](size_t bytes) -> void* {
        void* p = wsb + off;
        off = (off + bytes + 255) & ~(size_t)255;
        return p;
    };
    ushort_t* xb    = (ushort_t*)alloc(ME * 2);
    ushort_t* eb    = (ushort_t*)alloc(ME * 2);
    ushort_t* wq1T  = (ushort_t*)alloc((size_t)D * D * 2);
    ushort_t* wk1T  = (ushort_t*)alloc((size_t)D * D * 2);
    ushort_t* wv1T  = (ushort_t*)alloc((size_t)D * D * 2);
    ushort_t* wo1T  = (ushort_t*)alloc((size_t)D * D * 2);
    ushort_t* wq2T  = (ushort_t*)alloc((size_t)D * D * 2);
    ushort_t* wk2T  = (ushort_t*)alloc((size_t)D * D * 2);
    ushort_t* wv2T  = (ushort_t*)alloc((size_t)D * D * 2);
    ushort_t* wo2T  = (ushort_t*)alloc((size_t)D * D * 2);
    ushort_t* wf1T  = (ushort_t*)alloc((size_t)D * DFF * 2);
    ushort_t* wf2T  = (ushort_t*)alloc((size_t)DFF * D * 2);
    ushort_t* q1b   = (ushort_t*)alloc(ME * 2);
    ushort_t* k1b   = (ushort_t*)alloc(ME * 2);
    ushort_t* v1T   = (ushort_t*)alloc(ME * 2);   // [B][1024][S]
    ushort_t* ctx1b = (ushort_t*)alloc(ME * 2);
    float*    out1f = (float*)alloc(ME * 4);
    ushort_t* out1b = (ushort_t*)alloc(ME * 2);
    ushort_t* q2b   = (ushort_t*)alloc(ME * 2);
    ushort_t* k2b   = (ushort_t*)alloc(ME * 2);
    ushort_t* v2T   = (ushort_t*)alloc(ME * 2);
    ushort_t* ctx2b = (ushort_t*)alloc(ME * 2);
    float*    attn2f= (float*)alloc(ME * 4);
    float*    out2f = (float*)alloc(ME * 4);
    ushort_t* out2b = (ushort_t*)alloc(ME * 2);
    ushort_t* hidb  = (ushort_t*)alloc(HE * 2);
    float*    ffnf  = (float*)alloc(ME * 4);
    ushort_t* probs = (ushort_t*)alloc(PE * 2);   // big: reused by both MHAs

    // ---- converts / transposes ----
    cvt_f32_bf16_kernel<<<2048, 256, 0, stream>>>((const float4*)x,  (uint2*)xb, ME / 4);
    cvt_f32_bf16_kernel<<<2048, 256, 0, stream>>>((const float4*)enc,(uint2*)eb, ME / 4);
    transpose_cvt_kernel<<<dim3(D / 32, D / 32), 256, 0, stream>>>(wq1, wq1T, D, D);
    transpose_cvt_kernel<<<dim3(D / 32, D / 32), 256, 0, stream>>>(wk1, wk1T, D, D);
    transpose_cvt_kernel<<<dim3(D / 32, D / 32), 256, 0, stream>>>(wv1, wv1T, D, D);
    transpose_cvt_kernel<<<dim3(D / 32, D / 32), 256, 0, stream>>>(wo1, wo1T, D, D);
    transpose_cvt_kernel<<<dim3(D / 32, D / 32), 256, 0, stream>>>(wq2, wq2T, D, D);
    transpose_cvt_kernel<<<dim3(D / 32, D / 32), 256, 0, stream>>>(wk2, wk2T, D, D);
    transpose_cvt_kernel<<<dim3(D / 32, D / 32), 256, 0, stream>>>(wv2, wv2T, D, D);
    transpose_cvt_kernel<<<dim3(D / 32, D / 32), 256, 0, stream>>>(wo2, wo2T, D, D);
    transpose_cvt_kernel<<<dim3(DFF / 32, D / 32), 256, 0, stream>>>(wf1, wf1T, D, DFF);
    transpose_cvt_kernel<<<dim3(D / 32, DFF / 32), 256, 0, stream>>>(wf2, wf2T, DFF, D);

    const long long Z0 = 0;
    const long long vBstr = (long long)D * S;  // CbT per-batch stride

    // ---- MHA1: Q/K/V projections ----
    gemm_bf16_kernel<128><<<dim3(M / 128, D / 128, 1), 256, 0, stream>>>(
        xb, D, Z0, Z0, wq1T, D, Z0, Z0,
        (float*)nullptr, q1b, (ushort_t*)nullptr, D, Z0, Z0, bq1, 0, D, 1, 0);
    gemm_bf16_kernel<128><<<dim3(M / 128, D / 128, 1), 256, 0, stream>>>(
        xb, D, Z0, Z0, wk1T, D, Z0, Z0,
        (float*)nullptr, k1b, (ushort_t*)nullptr, D, Z0, Z0, bk1, 0, D, 1, 0);
    gemm_bf16_kernel<128><<<dim3(M / 128, D / 128, 1), 256, 0, stream>>>(
        xb, D, Z0, Z0, wv1T, D, Z0, Z0,
        (float*)nullptr, (ushort_t*)nullptr, v1T, D, Z0, Z0, bv1, 0, D, S, vBstr);

    // ---- MHA1: fused logits+softmax (mask = look_ahead [B,1,S,S]) ----
    attn_softmax_kernel<<<dim3(S / 16, 1, Bn * H), 256, 0, stream>>>(
        q1b, k1b, probs, (float*)nullptr, lam, 1, S, S, 0.125f);

    // ---- MHA1: ctx = P @ V  (A batched over b,h; C merges heads) ----
    gemm_bf16_kernel<64><<<dim3(S / 128, 1, Bn * H), 256, 0, stream>>>(
        probs, S, (long long)H * S * S, (long long)S * S,
        v1T, S, (long long)D * S, (long long)64 * S,
        (float*)nullptr, ctx1b, (ushort_t*)nullptr,
        D, (long long)S * D, (long long)64, (const float*)nullptr, 0, S, 1, 0);

    // ---- out1 = ctx1 @ wo1 + bo1 (f32 + bf16) ----
    gemm_bf16_kernel<128><<<dim3(M / 128, D / 128, 1), 256, 0, stream>>>(
        ctx1b, D, Z0, Z0, wo1T, D, Z0, Z0,
        out1f, out1b, (ushort_t*)nullptr, D, Z0, Z0, bo1, 0, D, 1, 0);

    // ---- MHA2 projections: Q from out1, K/V from enc ----
    gemm_bf16_kernel<128><<<dim3(M / 128, D / 128, 1), 256, 0, stream>>>(
        out1b, D, Z0, Z0, wq2T, D, Z0, Z0,
        (float*)nullptr, q2b, (ushort_t*)nullptr, D, Z0, Z0, bq2, 0, D, 1, 0);
    gemm_bf16_kernel<128><<<dim3(M / 128, D / 128, 1), 256, 0, stream>>>(
        eb, D, Z0, Z0, wk2T, D, Z0, Z0,
        (float*)nullptr, k2b, (ushort_t*)nullptr, D, Z0, Z0, bk2, 0, D, 1, 0);
    gemm_bf16_kernel<128><<<dim3(M / 128, D / 128, 1), 256, 0, stream>>>(
        eb, D, Z0, Z0, wv2T, D, Z0, Z0,
        (float*)nullptr, (ushort_t*)nullptr, v2T, D, Z0, Z0, bv2, 0, D, SE, vBstr);

    // ---- MHA2: fused logits+softmax; raw logits -> d_out ----
    attn_softmax_kernel<<<dim3(S / 16, 1, Bn * H), 256, 0, stream>>>(
        q2b, k2b, probs, logits, pam, 2, S, SE, 0.125f);

    // ---- MHA2: ctx2 = P @ V ----
    gemm_bf16_kernel<64><<<dim3(S / 128, 1, Bn * H), 256, 0, stream>>>(
        probs, SE, (long long)H * S * SE, (long long)S * SE,
        v2T, SE, (long long)D * SE, (long long)64 * SE,
        (float*)nullptr, ctx2b, (ushort_t*)nullptr,
        D, (long long)S * D, (long long)64, (const float*)nullptr, 0, SE, 1, 0);

    // ---- attn2 = ctx2 @ wo2 + bo2 (f32) ----
    gemm_bf16_kernel<128><<<dim3(M / 128, D / 128, 1), 256, 0, stream>>>(
        ctx2b, D, Z0, Z0, wo2T, D, Z0, Z0,
        attn2f, (ushort_t*)nullptr, (ushort_t*)nullptr, D, Z0, Z0, bo2, 0, D, 1, 0);

    // ---- out2 = LN2(attn2 + out1) ----
    layernorm_kernel<<<M, 256, 0, stream>>>(attn2f, out1f, ln2g, ln2b, out2f, out2b);

    // ---- FFN ----
    gemm_bf16_kernel<128><<<dim3(M / 128, DFF / 128, 1), 256, 0, stream>>>(
        out2b, D, Z0, Z0, wf1T, D, Z0, Z0,
        (float*)nullptr, hidb, (ushort_t*)nullptr, DFF, Z0, Z0, bf1, 1, D, 1, 0);
    gemm_bf16_kernel<128><<<dim3(M / 128, D / 128, 1), 256, 0, stream>>>(
        hidb, DFF, Z0, Z0, wf2T, DFF, Z0, Z0,
        ffnf, (ushort_t*)nullptr, (ushort_t*)nullptr, D, Z0, Z0, bf2, 0, DFF, 1, 0);

    // ---- out3 = LN3(ffn + out2) -> d_out ----
    layernorm_kernel<<<M, 256, 0, stream>>>(ffnf, out2f, ln3g, ln3b, out3,
                                            (ushort_t*)nullptr);
}
